// MultiHeadAttention_7292854469020
// MI455X (gfx1250) — matrix operand order repaired and checked by reading, not compiled, not run on hardware
//
#include <hip/hip_runtime.h>
#include <stdint.h>
#include <math.h>

typedef __attribute__((ext_vector_type(16))) __bf16 v16bf;
typedef __attribute__((ext_vector_type(8)))  float  v8f;

#define S_LEN  2048
#define HID_D  2048
#define NH_Q   16
#define NKV_H  4
#define HD_D   128
#define WIN    512
#define VROW   136   // padded row stride (ushorts) for K/V LDS tiles: 272B -> bank-conflict-free column gather

__device__ inline unsigned short f2bf(float f) {
  union { float f; unsigned u; } v; v.f = f;
  unsigned u = v.u;
  u += 0x7FFFu + ((u >> 16) & 1u);          // round-to-nearest-even
  return (unsigned short)(u >> 16);
}

__device__ inline v8f vzero8() { v8f v; for (int i = 0; i < 8; ++i) v[i] = 0.0f; return v; }

union FragAB { v16bf v; uint4 q[2]; unsigned short s[16]; };

// CDNA5 async global->LDS copy (ASYNCcnt-tracked, no VGPR round trip).
__device__ inline void async_b128(unsigned lds_addr, const void* gaddr) {
  asm volatile("global_load_async_to_lds_b128 %0, %1, off"
               :: "v"(lds_addr), "v"(gaddr) : "memory");
}
__device__ inline void wait_async0() {
  asm volatile("s_wait_asynccnt 0" ::: "memory");
}
__device__ inline unsigned lds_addr_of(const void* p) {
  return (unsigned)(uintptr_t)p;            // flat LDS addr truncates to LDS byte offset (ISA 10.2)
}

// ---------------- elementwise f32 -> bf16 ----------------
__global__ void mha_cvt_bf16(const float* __restrict__ src,
                             unsigned short* __restrict__ dst, int n) {
  int i = blockIdx.x * blockDim.x + threadIdx.x;
  if (i < n) dst[i] = f2bf(src[i]);
}

// ---------------- RoPE: f32 [s][h][128] -> bf16 [s][h][128] ----------------
__global__ void mha_rope(const float* __restrict__ src,
                         const float* __restrict__ cs,
                         const float* __restrict__ sn,
                         unsigned short* __restrict__ dst, int nheads) {
  int i = blockIdx.x * blockDim.x + threadIdx.x;
  int total = S_LEN * nheads * (HD_D / 2);
  if (i >= total) return;
  int d  = i & (HD_D / 2 - 1);
  int t  = i / (HD_D / 2);
  int hh = t % nheads;
  int s  = t / nheads;
  int base = (s * nheads + hh) * HD_D;
  float x1 = src[base + d], x2 = src[base + HD_D / 2 + d];
  float c = cs[s * (HD_D / 2) + d], sv = sn[s * (HD_D / 2) + d];
  dst[base + d]            = f2bf(x1 * c - x2 * sv);
  dst[base + HD_D / 2 + d] = f2bf(x1 * sv + x2 * c);
}

// ------- WMMA GEMM: C[M,N] = A[M,K] * B[N,K]^T, async double-buffered -------
__global__ __launch_bounds__(128)
void mha_gemm_bf16(const unsigned short* __restrict__ A,
                   const unsigned short* __restrict__ B,
                   float* __restrict__ C, int M, int N, int K) {
  __shared__ __align__(16) unsigned short lsA[2][64 * 32];
  __shared__ __align__(16) unsigned short lsB[2][64 * 32];
  const int tid  = threadIdx.x;
  const int wave = tid >> 5;
  const int lane = tid & 31;
  const int mr = lane & 15, g = lane >> 4;
  const int m0 = blockIdx.y * 64;
  const int n0 = blockIdx.x * 64;

  v8f acc[4];
  for (int j = 0; j < 4; ++j) acc[j] = vzero8();

  const int row = tid >> 1, half = tid & 1;
  auto stage = [&](int bufi, int k0) {
    const unsigned short* ga = A + (size_t)(m0 + row) * K + k0 + 16 * half;
    const unsigned short* gb = B + (size_t)(n0 + row) * K + k0 + 16 * half;
    unsigned la = lds_addr_of(&lsA[bufi][row * 32 + 16 * half]);
    unsigned lb = lds_addr_of(&lsB[bufi][row * 32 + 16 * half]);
    async_b128(la,      ga);
    async_b128(la + 16, ga + 8);
    async_b128(lb,      gb);
    async_b128(lb + 16, gb + 8);
    if (k0 + 32 < K) {                       // warm L2 for the tile after next
      __builtin_prefetch(ga + 32, 0, 1);
      __builtin_prefetch(gb + 32, 0, 1);
    }
  };

  stage(0, 0);
  wait_async0();
  __syncthreads();

  int buf = 0;
  for (int k0 = 0; k0 < K; k0 += 32) {
    if (k0 + 32 < K) stage(buf ^ 1, k0 + 32);   // overlap next slab with compute

    FragAB af, bfr[4];
    af.q[0] = *(const uint4*)&lsA[buf][(wave * 16 + mr) * 32 + 16 * g];
    af.q[1] = *(const uint4*)&lsA[buf][(wave * 16 + mr) * 32 + 16 * g + 8];
    for (int j = 0; j < 4; ++j) {
      bfr[j].q[0] = *(const uint4*)&lsB[buf][(j * 16 + mr) * 32 + 16 * g];
      bfr[j].q[1] = *(const uint4*)&lsB[buf][(j * 16 + mr) * 32 + 16 * g + 8];
    }
    for (int j = 0; j < 4; ++j)
      acc[j] = __builtin_amdgcn_wmma_f32_16x16x32_bf16(false, af.v, false, bfr[j].v,
                                                       (short)0, acc[j], false, false);

    wait_async0();
    __syncthreads();
    buf ^= 1;
  }

  for (int j = 0; j < 4; ++j)
    for (int r = 0; r < 8; ++r)
      C[(size_t)(m0 + wave * 16 + r + 8 * g) * N + n0 + j * 16 + mr] = acc[j][r];
}

// ---- Flash attention: 4 waves / 64 queries per block, shared K/V tiles -----
__global__ __launch_bounds__(128)
void mha_flash_attn(const unsigned short* __restrict__ Q,
                    const unsigned short* __restrict__ Km,
                    const unsigned short* __restrict__ Vm,
                    unsigned short* __restrict__ Out) {
  __shared__ __align__(16) unsigned short lsK[2][32 * VROW];
  __shared__ __align__(16) unsigned short lsV[2][32 * VROW];
  __shared__ __align__(16) unsigned short lsP[4][16 * 32];
  const int tid  = threadIdx.x;
  const int wave = tid >> 5;
  const int lane = tid & 31;
  const int mr = lane & 15, g = lane >> 4;
  const int qb0 = blockIdx.x * 64;
  const int q0  = qb0 + wave * 16;
  const int h   = blockIdx.y;
  const int hk  = h / (NH_Q / NKV_H);

  // Q tile 16x128 resident as 4 A-fragments
  FragAB qf[4];
  {
    const unsigned short* qrow = Q + ((size_t)(q0 + mr) * NH_Q + h) * HD_D;
    for (int c = 0; c < 4; ++c) {
      qf[c].q[0] = *(const uint4*)(qrow + 32 * c + 16 * g);
      qf[c].q[1] = *(const uint4*)(qrow + 32 * c + 16 * g + 8);
    }
  }

  v8f oacc[8];
  for (int t = 0; t < 8; ++t) oacc[t] = vzero8();
  float mrun[8], lrun[8];
  for (int r = 0; r < 8; ++r) { mrun[r] = -3.0e38f; lrun[r] = 0.0f; }

  int lo = qb0 - WIN; if (lo < 0) lo = 0; lo &= ~31;
  const int hi = ((qb0 + 63) >> 5) << 5;
  const float scale = 0.08838834764831845f;   // 1/sqrt(128)

  // staging: 128 threads cover 32 keys x 128 dims for both K and V
  const int krow_i = tid >> 2, kcol_i = (tid & 3) * 32;
  auto stage = [&](int bufi, int kt) {
    const unsigned short* ksrc = Km + ((size_t)(kt + krow_i) * NKV_H + hk) * HD_D + kcol_i;
    const unsigned short* vsrc = Vm + ((size_t)(kt + krow_i) * NKV_H + hk) * HD_D + kcol_i;
    unsigned lk = lds_addr_of(&lsK[bufi][krow_i * VROW + kcol_i]);
    unsigned lv = lds_addr_of(&lsV[bufi][krow_i * VROW + kcol_i]);
    for (int t = 0; t < 4; ++t) {
      async_b128(lk + 16 * t, ksrc + 8 * t);
      async_b128(lv + 16 * t, vsrc + 8 * t);
    }
  };

  stage(0, lo);
  wait_async0();
  __syncthreads();

  int buf = 0;
  for (int kt = lo; kt <= hi; kt += 32) {
    if (kt + 32 <= hi) stage(buf ^ 1, kt + 32);   // stream next key tile

    // S = Q * K^T : two 16-key subtiles, 4 chained wmma each
    v8f sacc[2];
    sacc[0] = vzero8(); sacc[1] = vzero8();
    for (int t = 0; t < 2; ++t)
      for (int c = 0; c < 4; ++c) {
        FragAB kf;
        kf.q[0] = *(const uint4*)&lsK[buf][(16 * t + mr) * VROW + 32 * c + 16 * g];
        kf.q[1] = *(const uint4*)&lsK[buf][(16 * t + mr) * VROW + 32 * c + 16 * g + 8];
        sacc[t] = __builtin_amdgcn_wmma_f32_16x16x32_bf16(false, qf[c].v, false, kf.v,
                                                          (short)0, sacc[t], false, false);
      }

    // online softmax; row m = r + 8g spans the lane's 16-lane group
    for (int r = 0; r < 8; ++r) {
      const int i  = q0 + r + 8 * g;
      const int j0 = kt + mr, j1 = kt + 16 + mr;
      float v0 = sacc[0][r] * scale;
      float v1 = sacc[1][r] * scale;
      if (j0 > i || (i - j0) > WIN) v0 = -1.0e30f;   // causal + sliding window
      if (j1 > i || (i - j1) > WIN) v1 = -1.0e30f;
      float t = fmaxf(v0, v1);
      for (int msk = 8; msk >= 1; msk >>= 1)
        t = fmaxf(t, __shfl_xor(t, msk, 16));
      const float mnew  = fmaxf(mrun[r], t);
      const float alpha = __expf(mrun[r] - mnew);
      const float p0 = __expf(v0 - mnew);
      const float p1 = __expf(v1 - mnew);
      float srow = p0 + p1;
      for (int msk = 8; msk >= 1; msk >>= 1)
        srow += __shfl_xor(srow, msk, 16);
      lrun[r] = lrun[r] * alpha + srow;
      mrun[r] = mnew;
      for (int t2 = 0; t2 < 8; ++t2) oacc[t2][r] *= alpha;
      lsP[wave][(r + 8 * g) * 32 + mr]      = f2bf(p0);
      lsP[wave][(r + 8 * g) * 32 + 16 + mr] = f2bf(p1);
    }

    // P as A-fragment (per-wave LDS region; DS ops are in-order per wave)
    FragAB pf;
    pf.q[0] = *(const uint4*)&lsP[wave][mr * 32 + 16 * g];
    pf.q[1] = *(const uint4*)&lsP[wave][mr * 32 + 16 * g + 8];

    // O += P * V : 8 wmma across HD; padded VROW makes the gather conflict-free
    for (int dc = 0; dc < 8; ++dc) {
      FragAB vf;
      for (int e = 0; e < 16; ++e)
        vf.s[e] = lsV[buf][(16 * g + e) * VROW + dc * 16 + mr];
      oacc[dc] = __builtin_amdgcn_wmma_f32_16x16x32_bf16(false, pf.v, false, vf.v,
                                                         (short)0, oacc[dc], false, false);
    }

    wait_async0();
    __syncthreads();
    buf ^= 1;
  }

  for (int r = 0; r < 8; ++r) {
    const float inv = 1.0f / lrun[r];
    unsigned short* orow = Out + (size_t)(q0 + r + 8 * g) * (NH_Q * HD_D) + h * HD_D;
    for (int dc = 0; dc < 8; ++dc)
      orow[dc * 16 + mr] = f2bf(oacc[dc][r] * inv);
  }
}

// ---------------------------------------------------------------------------
extern "C" void kernel_launch(void* const* d_in, const int* in_sizes, int n_in,
                              void* d_out, int out_size, void* d_ws, size_t ws_size,
                              hipStream_t stream) {
  const float* x  = (const float*)d_in[0];
  const float* wq = (const float*)d_in[1];
  const float* wk = (const float*)d_in[2];
  const float* wv = (const float*)d_in[3];
  const float* wo = (const float*)d_in[4];
  // d_in[5] = attention_mask (recomputed inline)
  const float* fc = (const float*)d_in[6];
  const float* fs = (const float*)d_in[7];
  float* out = (float*)d_out;

  char* ws = (char*)d_ws;
  size_t off = 0;
  auto carve = [&](size_t bytes) -> void* {
    void* p = ws + off; off += (bytes + 255) & ~(size_t)255; return p;
  };
  const size_t NX  = (size_t)S_LEN * HID_D;          // 4M
  const size_t NKW = (size_t)(NKV_H * HD_D) * HID_D; // 1M
  unsigned short* xb   = (unsigned short*)carve(NX  * 2);
  unsigned short* wqb  = (unsigned short*)carve(NX  * 2);
  unsigned short* wkb  = (unsigned short*)carve(NKW * 2);
  unsigned short* wvb  = (unsigned short*)carve(NKW * 2);
  unsigned short* wob  = (unsigned short*)carve(NX  * 2);
  float*          qf32 = (float*)carve(NX  * 4);
  float*          kf32 = (float*)carve((size_t)S_LEN * NKV_H * HD_D * 4);
  float*          vf32 = (float*)carve((size_t)S_LEN * NKV_H * HD_D * 4);
  unsigned short* qb   = (unsigned short*)carve(NX * 2);
  unsigned short* kb   = (unsigned short*)carve((size_t)S_LEN * NKV_H * HD_D * 2);
  unsigned short* vb   = (unsigned short*)carve((size_t)S_LEN * NKV_H * HD_D * 2);
  unsigned short* attb = (unsigned short*)carve(NX * 2);
  (void)ws_size; (void)in_sizes; (void)n_in; (void)out_size;

  const int T = 256;
  mha_cvt_bf16<<<(int)((NX  + T - 1) / T), T, 0, stream>>>(x,  xb,  (int)NX);
  mha_cvt_bf16<<<(int)((NX  + T - 1) / T), T, 0, stream>>>(wq, wqb, (int)NX);
  mha_cvt_bf16<<<(int)((NKW + T - 1) / T), T, 0, stream>>>(wk, wkb, (int)NKW);
  mha_cvt_bf16<<<(int)((NKW + T - 1) / T), T, 0, stream>>>(wv, wvb, (int)NKW);
  mha_cvt_bf16<<<(int)((NX  + T - 1) / T), T, 0, stream>>>(wo, wob, (int)NX);

  // projections: q = x*wq^T, k = x*wk^T, v = x*wv^T
  dim3 gq(HID_D / 64, S_LEN / 64);
  mha_gemm_bf16<<<gq, 128, 0, stream>>>(xb, wqb, qf32, S_LEN, HID_D, HID_D);
  dim3 gkv((NKV_H * HD_D) / 64, S_LEN / 64);
  mha_gemm_bf16<<<gkv, 128, 0, stream>>>(xb, wkb, kf32, S_LEN, NKV_H * HD_D, HID_D);
  mha_gemm_bf16<<<gkv, 128, 0, stream>>>(xb, wvb, vf32, S_LEN, NKV_H * HD_D, HID_D);

  // RoPE(q,k) -> bf16 ; v -> bf16
  int nq = S_LEN * NH_Q  * (HD_D / 2);
  int nk = S_LEN * NKV_H * (HD_D / 2);
  mha_rope<<<(nq + T - 1) / T, T, 0, stream>>>(qf32, fc, fs, qb, NH_Q);
  mha_rope<<<(nk + T - 1) / T, T, 0, stream>>>(kf32, fc, fs, kb, NKV_H);
  int nv = S_LEN * NKV_H * HD_D;
  mha_cvt_bf16<<<(nv + T - 1) / T, T, 0, stream>>>(vf32, vb, nv);

  // windowed causal flash attention (64 queries x 1 head per block)
  dim3 gfa(S_LEN / 64, NH_Q);
  mha_flash_attn<<<gfa, 128, 0, stream>>>(qb, kb, vb, attb);

  // output projection -> f32 d_out
  dim3 go(HID_D / 64, S_LEN / 64);
  mha_gemm_bf16<<<go, 128, 0, stream>>>(attb, wob, out, S_LEN, HID_D, HID_D);
}
